// EdgeAwareAttentionFusion_58841051955655
// MI455X (gfx1250) — matrix operand order repaired and checked by reading, not compiled, not run on hardware
//
#include <hip/hip_runtime.h>
#include <hip/hip_bf16.h>

typedef __attribute__((ext_vector_type(16))) __bf16 bf16x16;
typedef __attribute__((ext_vector_type(8)))  __bf16 bf16x8;
typedef __attribute__((ext_vector_type(8)))  float  f32x8;

#define EPS_ 1e-5f
#define QSTR 53   // padded stride for Q/K/V rows in LDS (odd -> conflict-friendly)

__device__ __forceinline__ unsigned short f2bf(float f) {
    unsigned int u = __float_as_uint(f);
    u += 0x7FFFu + ((u >> 16) & 1u);   // round-to-nearest-even
    return (unsigned short)(u >> 16);
}

// ---------------------------------------------------------------------------
// K0: convert wq, wk, wv, wp (each 256x256 fp32) to bf16 into ws
// ---------------------------------------------------------------------------
__global__ void k_cvt_w(const float* __restrict__ wq, const float* __restrict__ wk,
                        const float* __restrict__ wv, const float* __restrict__ wp,
                        unsigned short* __restrict__ out) {
    int i = blockIdx.x * blockDim.x + threadIdx.x;     // 0..262143
    const float* src[4] = {wq, wk, wv, wp};
    int w = i >> 16, j = i & 65535;
    out[i] = f2bf(src[w][j]);
}

// ---------------------------------------------------------------------------
// K1: mega window kernel: gather+BN -> 3x WMMA projection GEMMs -> attention
//     one workgroup (256 threads / 8 waves) per window (4096 windows)
// LDS layout (bytes):
//   [0      ) Ae   64x256 bf16 (32768)   -- x_edge window, [s][c]
//   [32768  ) Av   64x256 bf16 (32768)   -- x_vit window
//   [65536  ) Af   64x256 bf16 (32768)   -- BN(xe+xv) window
//   [98304  ) Wc   256x32 bf16 (16384)   -- weight K-chunk
//   [114688 ) Qs   256x53 f32  (54272)
//   [168960 ) Ks   256x53 f32  (54272)
//   [223232 ) Vs   256x53 f32  (54272)   total 277504
// ---------------------------------------------------------------------------
__global__ __launch_bounds__(256, 1) void k_win_attn(
    const float* __restrict__ xe, const float* __restrict__ xv,
    const float* __restrict__ g1, const float* __restrict__ b1,
    const float* __restrict__ m1, const float* __restrict__ v1,
    const unsigned short* __restrict__ wbf,   // [3][256][256] bf16: q,k,v
    float* __restrict__ xsp, float* __restrict__ imp_raw)
{
    extern __shared__ char smem[];
    unsigned short* Ae = (unsigned short*)(smem);
    unsigned short* Av = (unsigned short*)(smem + 32768);
    unsigned short* Af = (unsigned short*)(smem + 65536);
    unsigned short* Wc = (unsigned short*)(smem + 98304);
    float* Qs = (float*)(smem + 114688);
    float* Ks = (float*)(smem + 168960);
    float* Vs = (float*)(smem + 223232);

    const int t  = threadIdx.x;
    const int n  = blockIdx.x;          // window id = (b*16 + wh)*16 + ww
    const int b  = n >> 8;
    const int wh = (n >> 4) & 15;
    const int ww = n & 15;
    const int h0 = wh * 7, w0 = ww * 7;

    // ---- gather window + BN fuse, store bf16 [s][c] ----
    for (int idx = t; idx < 256 * 49; idx += 256) {
        int c = idx / 49, s = idx - c * 49;
        int sh = s / 7, sw = s - sh * 7;
        long off = (((long)(b * 256 + c)) * 112 + (h0 + sh)) * 112 + (w0 + sw);
        float e = xe[off], v = xv[off];
        float sc = g1[c] * rsqrtf(v1[c] + EPS_);
        float f  = (e + v - m1[c]) * sc + b1[c];
        Ae[s * 256 + c] = f2bf(e);
        Av[s * 256 + c] = f2bf(v);
        Af[s * 256 + c] = f2bf(f);
    }
    // zero-pad rows s=49..63
    for (int idx = t; idx < 15 * 256; idx += 256) {
        int s = 49 + (idx >> 8);
        int c = idx & 255;
        Ae[s * 256 + c] = 0; Av[s * 256 + c] = 0; Af[s * 256 + c] = 0;
    }

    const int lane = t & 31, wid = t >> 5;
    const int half = lane >> 4, lm = lane & 15;

    // ---- 3 projection GEMMs: out[256 o][64 s] = W[256x256] * X[256 c][64 s] ----
    for (int p = 0; p < 3; ++p) {
        const unsigned short* Wg = wbf + p * 65536;
        const unsigned short* A  = (p == 0) ? Ae : ((p == 1) ? Av : Af);
        float* O = (p == 0) ? Qs : ((p == 1) ? Ks : Vs);

        f32x8 acc[2][4] = {};
        for (int kc = 0; kc < 8; ++kc) {
            const int k0 = kc * 32;
            __syncthreads();
            { // stage weight chunk Wc[256][32]
                const unsigned short* src = Wg + t * 256 + k0;
                unsigned short* dst = Wc + t * 32;
                *(uint4*)(dst +  0) = *(const uint4*)(src +  0);
                *(uint4*)(dst +  8) = *(const uint4*)(src +  8);
                *(uint4*)(dst + 16) = *(const uint4*)(src + 16);
                *(uint4*)(dst + 24) = *(const uint4*)(src + 24);
                if (kc < 7) __builtin_prefetch(src + 32, 0, 1);
            }
            __syncthreads();

            bf16x16 afr[2];
#pragma unroll
            for (int i = 0; i < 2; ++i) {
                int o = (wid * 2 + i) * 16 + lm;
                bf16x8 lo = *(const bf16x8*)(Wc + o * 32 + half * 16);
                bf16x8 hi = *(const bf16x8*)(Wc + o * 32 + half * 16 + 8);
#pragma unroll
                for (int e = 0; e < 8; ++e) { afr[i][e] = lo[e]; afr[i][8 + e] = hi[e]; }
            }
#pragma unroll
            for (int j = 0; j < 4; ++j) {
                int s = j * 16 + lm;
                bf16x16 bfr = *(const bf16x16*)(A + s * 256 + k0 + half * 16);
#pragma unroll
                for (int i = 0; i < 2; ++i)
                    acc[i][j] = __builtin_amdgcn_wmma_f32_16x16x32_bf16(
                        false, afr[i], false, bfr, (short)0, acc[i][j], false, false);
            }
        }
        // epilogue -> LDS [o][QSTR], ReLU on V projection
#pragma unroll
        for (int i = 0; i < 2; ++i)
#pragma unroll
            for (int j = 0; j < 4; ++j)
#pragma unroll
                for (int r = 0; r < 8; ++r) {
                    int o = (wid * 2 + i) * 16 + r + 8 * half;
                    int s = j * 16 + lm;
                    float val = acc[i][j][r];
                    if (p == 2 && val < 0.f) val = 0.f;
                    if (s < 49) O[o * QSTR + s] = val;
                }
    }
    __syncthreads();

    // ---- attention: 4 heads x 49 tokens, dim 64 (VALU; ~10% of FLOPs) ----
    float* sc  = (float*)(smem);          // 196*50 f32 over dead Ae/Av
    float* red = (float*)(smem + 98304);  // dead Wc region
    if (t == 0) red[0] = 0.f;
    __syncthreads();

    if (t < 196) {
        const int h = t / 49, q = t - h * 49;
        float qv[64];
#pragma unroll
        for (int d = 0; d < 64; ++d) qv[d] = Qs[(h * 64 + d) * QSTR + q];

        float mx = -1e30f;
        for (int k = 0; k < 49; ++k) {
            float s = 0.f;
#pragma unroll
            for (int d = 0; d < 64; ++d) s += qv[d] * Ks[(h * 64 + d) * QSTR + k];
            s *= 0.125f;                  // DH^-0.5
            sc[t * 50 + k] = s;
            mx = fmaxf(mx, s);
        }
        float sum = 0.f;
        for (int k = 0; k < 49; ++k) { float e = __expf(sc[t * 50 + k] - mx); sc[t * 50 + k] = e; sum += e; }
        float inv = 1.f / sum, rs = 0.f;
        for (int k = 0; k < 49; ++k) { float a = sc[t * 50 + k] * inv; sc[t * 50 + k] = a; rs += fabsf(a); }
        atomicAdd(red, rs);

        for (int d = 0; d < 64; ++d) {
            float o = 0.f;
            for (int k = 0; k < 49; ++k) o += sc[t * 50 + k] * Vs[(h * 64 + d) * QSTR + k];
            xsp[((long)(n * 256 + h * 64 + d)) * 49 + q] = o;
        }
    }
    __syncthreads();
    if (t == 0) imp_raw[n] = red[0] * (1.0f / (4.0f * 49.0f * 49.0f));
}

// ---------------------------------------------------------------------------
// K2: global max over imp_raw[4096]
// ---------------------------------------------------------------------------
__global__ void k_impmax(const float* __restrict__ imp, float* __restrict__ gmax) {
    __shared__ float r[256];
    float m = -1e30f;
    for (int i = threadIdx.x; i < 4096; i += 256) m = fmaxf(m, imp[i]);
    r[threadIdx.x] = m; __syncthreads();
    for (int s = 128; s > 0; s >>= 1) {
        if (threadIdx.x < s) r[threadIdx.x] = fmaxf(r[threadIdx.x], r[threadIdx.x + s]);
        __syncthreads();
    }
    if (threadIdx.x == 0) gmax[0] = r[0];
}

// ---------------------------------------------------------------------------
// K3: blend x = imp*x_sp + (1-imp)*vw (vw recomputed), fold to NCHW
// ---------------------------------------------------------------------------
__global__ void k_blend(const float* __restrict__ xsp, const float* __restrict__ imp_raw,
                        const float* __restrict__ gmax,
                        const float* __restrict__ xe, const float* __restrict__ xv,
                        const float* __restrict__ g1, const float* __restrict__ b1,
                        const float* __restrict__ m1, const float* __restrict__ v1,
                        float* __restrict__ xo) {
    long e = (long)blockIdx.x * blockDim.x + threadIdx.x;
    if (e >= 51380224L) return;
    int w = (int)(e % 112); long r = e / 112;
    int h = (int)(r % 112); r /= 112;
    int c = (int)(r % 256); int b = (int)(r / 256);
    int n = ((b * 16) + h / 7) * 16 + w / 7;
    int s = (h % 7) * 7 + (w % 7);
    float imp = imp_raw[n] / (gmax[0] + 1e-6f);
    float scl = g1[c] * rsqrtf(v1[c] + EPS_);
    float f = (xe[e] + xv[e] - m1[c]) * scl + b1[c];
    xo[e] = imp * xsp[((long)(n * 256 + c)) * 49 + s] + (1.f - imp) * f;
}

// ---------------------------------------------------------------------------
// K4: depthwise 3x3 + BN2 + SiLU + fused global-average pool
//     one workgroup per (b,c) plane
// ---------------------------------------------------------------------------
__global__ __launch_bounds__(256) void k_dwconv(
    const float* __restrict__ x, const float* __restrict__ dww,
    const float* __restrict__ g2, const float* __restrict__ b2,
    const float* __restrict__ m2, const float* __restrict__ v2,
    float* __restrict__ y, float* __restrict__ pool) {
    __shared__ float tile[114 * 114];
    __shared__ float rs[256];
    const int bc = blockIdx.x;           // b*256 + c
    const int c = bc & 255;
    const float* xp = x + (long)bc * 12544;
    for (int i = threadIdx.x; i < 114 * 114; i += 256) {
        int hh = i / 114 - 1, wc = i % 114 - 1;
        tile[i] = (hh >= 0 && hh < 112 && wc >= 0 && wc < 112) ? xp[hh * 112 + wc] : 0.f;
    }
    __syncthreads();
    float kk[9];
#pragma unroll
    for (int i = 0; i < 9; ++i) kk[i] = dww[c * 9 + i];
    const float scl = g2[c] * rsqrtf(v2[c] + EPS_);
    const float mm = m2[c], bb = b2[c];
    float acc = 0.f;
    float* yp = y + (long)bc * 12544;
    for (int i = threadIdx.x; i < 12544; i += 256) {
        int h = i / 112, w = i % 112;
        const float* tp = tile + h * 114 + w;
        float s = tp[0] * kk[0] + tp[1] * kk[1] + tp[2] * kk[2]
                + tp[114] * kk[3] + tp[115] * kk[4] + tp[116] * kk[5]
                + tp[228] * kk[6] + tp[229] * kk[7] + tp[230] * kk[8];
        s = (s - mm) * scl + bb;
        s = s / (1.f + __expf(-s));      // SiLU
        yp[i] = s;
        acc += s;
    }
    rs[threadIdx.x] = acc; __syncthreads();
    for (int s2 = 128; s2 > 0; s2 >>= 1) {
        if (threadIdx.x < s2) rs[threadIdx.x] += rs[threadIdx.x + s2];
        __syncthreads();
    }
    if (threadIdx.x == 0) pool[bc] = rs[0] * (1.f / 12544.f);
}

// ---------------------------------------------------------------------------
// K5: ECA: 1D conv (k=3, pad=1) over channels + sigmoid
// ---------------------------------------------------------------------------
__global__ void k_eca(const float* __restrict__ pool, const float* __restrict__ ew,
                      float* __restrict__ sig) {
    int b = blockIdx.x, c = threadIdx.x;
    float pm1 = (c > 0)   ? pool[b * 256 + c - 1] : 0.f;
    float p0  =             pool[b * 256 + c];
    float pp1 = (c < 255) ? pool[b * 256 + c + 1] : 0.f;
    float a = ew[0] * pm1 + ew[1] * p0 + ew[2] * pp1;
    sig[b * 256 + c] = 1.f / (1.f + __expf(-a));
}

// ---------------------------------------------------------------------------
// K6: output projection: out[b,o,hw] = x[b,o,hw] + sum_c wp[o,c]*(y[b,c,hw]*sig[b,c])
//     WMMA bf16, M=256, N=64 per block, K=256; grid (196 hw-tiles, 16 b)
// ---------------------------------------------------------------------------
__global__ __launch_bounds__(256, 2) void k_outgemm(
    const unsigned short* __restrict__ wpbf,
    const float* __restrict__ y, const float* __restrict__ sig,
    const float* __restrict__ x, float* __restrict__ out) {
    __shared__ __align__(64) unsigned short Wc[256 * 32];
    __shared__ __align__(64) unsigned short Yt[64 * 32];
    const int t = threadIdx.x;
    const int b = blockIdx.y;
    const int hw0 = blockIdx.x * 64;
    const int lane = t & 31, wid = t >> 5, half = lane >> 4, lm = lane & 15;

    f32x8 acc[2][4] = {};
    for (int kc = 0; kc < 8; ++kc) {
        const int k0 = kc * 32;
        __syncthreads();
        { // weight chunk
            const unsigned short* src = wpbf + t * 256 + k0;
            unsigned short* dst = Wc + t * 32;
            *(uint4*)(dst +  0) = *(const uint4*)(src +  0);
            *(uint4*)(dst +  8) = *(const uint4*)(src +  8);
            *(uint4*)(dst + 16) = *(const uint4*)(src + 16);
            *(uint4*)(dst + 24) = *(const uint4*)(src + 24);
            if (kc < 7) __builtin_prefetch(src + 32, 0, 1);
        }
        { // activation chunk: y scaled by sigmoid, transposed to Yt[col][k]
            int k = t >> 3, col = (t & 7) * 8;
            const float* yp = y + ((long)(b * 256 + k0 + k)) * 12544 + hw0 + col;
            float sg = sig[b * 256 + k0 + k];
#pragma unroll
            for (int i = 0; i < 8; ++i) Yt[(col + i) * 32 + k] = f2bf(yp[i] * sg);
        }
        __syncthreads();

        bf16x16 afr[2];
#pragma unroll
        for (int i = 0; i < 2; ++i) {
            int o = (wid * 2 + i) * 16 + lm;
            bf16x8 lo = *(const bf16x8*)(Wc + o * 32 + half * 16);
            bf16x8 hi = *(const bf16x8*)(Wc + o * 32 + half * 16 + 8);
#pragma unroll
            for (int e = 0; e < 8; ++e) { afr[i][e] = lo[e]; afr[i][8 + e] = hi[e]; }
        }
#pragma unroll
        for (int j = 0; j < 4; ++j) {
            int s = j * 16 + lm;
            bf16x16 bfr = *(const bf16x16*)(Yt + s * 32 + half * 16);
#pragma unroll
            for (int i = 0; i < 2; ++i)
                acc[i][j] = __builtin_amdgcn_wmma_f32_16x16x32_bf16(
                    false, afr[i], false, bfr, (short)0, acc[i][j], false, false);
        }
    }
    // epilogue: residual add + store
#pragma unroll
    for (int i = 0; i < 2; ++i)
#pragma unroll
        for (int j = 0; j < 4; ++j)
#pragma unroll
            for (int r = 0; r < 8; ++r) {
                int o = (wid * 2 + i) * 16 + r + 8 * half;
                int col = j * 16 + lm;
                long off = ((long)(b * 256 + o)) * 12544 + hw0 + col;
                out[off] = x[off] + acc[i][j][r];
            }
}

// ---------------------------------------------------------------------------
extern "C" void kernel_launch(void* const* d_in, const int* in_sizes, int n_in,
                              void* d_out, int out_size, void* d_ws, size_t ws_size,
                              hipStream_t stream) {
    (void)in_sizes; (void)n_in; (void)out_size; (void)ws_size;
    const float* xe  = (const float*)d_in[0];
    const float* xv  = (const float*)d_in[1];
    const float* g1  = (const float*)d_in[2];
    const float* b1  = (const float*)d_in[3];
    const float* m1  = (const float*)d_in[4];
    const float* v1  = (const float*)d_in[5];
    const float* wq  = (const float*)d_in[6];
    const float* wk  = (const float*)d_in[7];
    const float* wv  = (const float*)d_in[8];
    const float* dww = (const float*)d_in[9];
    const float* g2  = (const float*)d_in[10];
    const float* b2  = (const float*)d_in[11];
    const float* m2  = (const float*)d_in[12];
    const float* v2  = (const float*)d_in[13];
    const float* ew  = (const float*)d_in[14];
    const float* wp  = (const float*)d_in[15];

    char* ws = (char*)d_ws;
    unsigned short* wbf  = (unsigned short*)(ws);            // 4*65536 bf16 = 512KB
    float* imp  = (float*)(ws + 524288);                     // 4096 f32
    float* gmax = (float*)(ws + 540672);
    float* pool = (float*)(ws + 540928);                     // 4096 f32
    float* sig  = (float*)(ws + 557312);                     // 4096 f32
    float* xsp  = (float*)(ws + 573696);                     // 4096*256*49 f32 (196MB)
    float* y_   = xsp;                                       // reuse after blend consumes xsp
    float* x_   = (float*)(ws + 573696 + 205520896UL);       // 196MB
    float* out  = (float*)d_out;

    k_cvt_w<<<1024, 256, 0, stream>>>(wq, wk, wv, wp, wbf);

    const size_t smem = 277504;
    (void)hipFuncSetAttribute(reinterpret_cast<const void*>(k_win_attn),
                              hipFuncAttributeMaxDynamicSharedMemorySize, (int)smem);
    k_win_attn<<<4096, 256, smem, stream>>>(xe, xv, g1, b1, m1, v1, wbf, xsp, imp);

    k_impmax<<<1, 256, 0, stream>>>(imp, gmax);
    k_blend<<<200704, 256, 0, stream>>>(xsp, imp, gmax, xe, xv, g1, b1, m1, v1, x_);
    k_dwconv<<<4096, 256, 0, stream>>>(x_, dww, g2, b2, m2, v2, y_, pool);
    k_eca<<<16, 256, 0, stream>>>(pool, ew, sig);
    k_outgemm<<<dim3(196, 16), 256, 0, stream>>>(wbf + 3 * 65536, y_, sig, x_, out);
}
